// AttentionPooling_vectorized_65678639890659
// MI455X (gfx1250) — matrix operand order repaired and checked by reading, not compiled, not run on hardware
//
#include <hip/hip_runtime.h>
#include <hip/hip_bf16.h>

// ---------------- problem constants (match reference) ----------------
#define B_   8
#define SL_  512
#define H_   768
#define NS_  256
#define W_   32
#define NH_  12
#define HD_  64
#define FFN_ 3072
#define N_   (B_ * NS_)   // 2048 spans
#define LN_EPS_ 1e-5f

typedef __attribute__((ext_vector_type(16))) _Float16 v16h;
typedef __attribute__((ext_vector_type(8)))  float    v8f;

// ---------------- TDM (Tensor Data Mover) availability ----------------
#ifdef __has_builtin
#if __has_builtin(__builtin_amdgcn_tensor_load_to_lds)
#define HAVE_TDM_ 1
#endif
#endif
#ifndef HAVE_TDM_
#define HAVE_TDM_ 0
#endif

#if __has_include(<hip/amd_detail/amd_gfx1250_TDM.h>)
#define TDM_SIX_ARGS_ 1
#else
#define TDM_SIX_ARGS_ 0
#endif

#if HAVE_TDM_
typedef __attribute__((ext_vector_type(4))) unsigned int u32x4_;
typedef __attribute__((ext_vector_type(8))) int          i32x8_;
typedef __attribute__((ext_vector_type(4))) int          i32x4_;

// Issue one 2-D TDM tile load: global (row-major, rowStride halves) -> LDS.
// Tile = tile0 halves x tile1 rows, f16 elements, optional LDS row padding
// (pad_interval code 3 = 16 DWORDs = 64B rows, pad_amount code 3 = 4 DWORDs
//  = 16B pad) -> matches a [64][32+8] f16 LDS tile exactly.
// D# packing per CDNA5 ISA.
__device__ __forceinline__ void tdm_load_tile_f16(unsigned int lds_off,
                                                  const void* gaddr,
                                                  unsigned int tile0,
                                                  unsigned int tile1,
                                                  unsigned int rowStride,
                                                  int pad_en) {
  unsigned long long ga = (unsigned long long)(uintptr_t)gaddr;
  u32x4_ g0;
  g0[0] = 1u;                                   // count=1, user descriptor
  g0[1] = lds_off;                              // lds_addr (bytes)
  g0[2] = (unsigned int)ga;                     // global_addr[31:0]
  g0[3] = (unsigned int)((ga >> 32) & 0x01FFFFFFu) | (2u << 30);  // type=2

  unsigned int w0 = (1u << 16);                 // data_size=1 -> 2 bytes
  if (pad_en) w0 |= (1u << 20) | (3u << 22) | (3u << 25);  // pad 16DW int, 4DW amt
  i32x8_ g1;
  g1[0] = (int)w0;                              // wg_mask=0 | flags
  g1[1] = (int)((tile0 & 0xFFFFu) << 16);       // tensor_dim0[15:0] (== tile0)
  g1[2] = (int)((tile1 & 0xFFFFu) << 16);       // dim0[31:16]=0 | tensor_dim1[15:0]
  g1[3] = (int)((tile0 & 0xFFFFu) << 16);       // dim1[31:16]=0 | tile_dim0
  g1[4] = (int)(tile1 & 0xFFFFu);               // tile_dim1 | tile_dim2=0
  g1[5] = (int)rowStride;                       // tensor_dim0_stride[31:0]
  g1[6] = 0;                                    // stride0[47:32] | dim1_stride[15:0]
  g1[7] = 0;

  i32x4_ g2 = {0, 0, 0, 0};
  i32x4_ g3 = {0, 0, 0, 0};
#if TDM_SIX_ARGS_
  i32x8_ gx = {0, 0, 0, 0, 0, 0, 0, 0};
  __builtin_amdgcn_tensor_load_to_lds(g0, g1, g2, g3, gx, 0);
#else
  __builtin_amdgcn_tensor_load_to_lds(g0, g1, g2, g3, 0);
#endif
}
#endif  // HAVE_TDM_

// =====================================================================
// prep: token_reps + sinusoidal positional encoding -> f16
// =====================================================================
__global__ void prep_tokpe_kernel(const float* __restrict__ tok,
                                  _Float16* __restrict__ out, int total) {
  for (int i = blockIdx.x * blockDim.x + threadIdx.x; i < total;
       i += gridDim.x * blockDim.x) {
    int c = i % H_;
    int p = (i / H_) % SL_;
    float dv  = __expf(-9.210340371976184f * (float)(c & ~1) / (float)H_);
    float ang = (float)p * dv;
    float pe  = (c & 1) ? __cosf(ang) : __sinf(ang);
    out[i] = (_Float16)(tok[i] + pe);
  }
}

// =====================================================================
// prep: convert weights (Wk, Wv, Wo, W1, W2) to f16, row-major (out, in)
// =====================================================================
__global__ void prep_weights_kernel(const float* __restrict__ in_proj_w,
                                    const float* __restrict__ out_proj_w,
                                    const float* __restrict__ ffn_w1,
                                    const float* __restrict__ ffn_w2,
                                    _Float16* __restrict__ wk,
                                    _Float16* __restrict__ wv,
                                    _Float16* __restrict__ wo,
                                    _Float16* __restrict__ w1,
                                    _Float16* __restrict__ w2) {
  const int HH = H_ * H_;
  const int FH = FFN_ * H_;
  const int total = 3 * HH + 2 * FH;
  for (int i = blockIdx.x * blockDim.x + threadIdx.x; i < total;
       i += gridDim.x * blockDim.x) {
    if (i < HH)                wk[i]               = (_Float16)in_proj_w[HH + i];
    else if (i < 2 * HH)       wv[i - HH]          = (_Float16)in_proj_w[2 * HH + (i - HH)];
    else if (i < 3 * HH)       wo[i - 2 * HH]      = (_Float16)out_proj_w[i - 2 * HH];
    else if (i < 3 * HH + FH)  w1[i - 3 * HH]      = (_Float16)ffn_w1[i - 3 * HH];
    else                       w2[i - 3 * HH - FH] = (_Float16)ffn_w2[i - 3 * HH - FH];
  }
}

// =====================================================================
// q = Wq @ q0 + bq  (tiny GEMV, 768x768)
// =====================================================================
__global__ void qproj_kernel(const float* __restrict__ in_proj_w,
                             const float* __restrict__ in_proj_b,
                             const float* __restrict__ q0,
                             float* __restrict__ qout) {
  int i = blockIdx.x * blockDim.x + threadIdx.x;
  if (i < H_) {
    float s = in_proj_b[i];
    for (int k = 0; k < H_; ++k) s += in_proj_w[i * H_ + k] * q0[k];
    qout[i] = s;
  }
}

// =====================================================================
// Fused per-span kernel: gather + (K,V) WMMA projection + masked softmax
// attention -> ctx (f16).  One workgroup (8 waves) per span.
// Weights stream from global (L2-resident, reused by all 2048 blocks).
// =====================================================================
__global__ __launch_bounds__(256)
void span_attn_kernel(const _Float16* __restrict__ tokpe,
                      const int*      __restrict__ span_ids,
                      const float*    __restrict__ qf,
                      const _Float16* __restrict__ wk,
                      const _Float16* __restrict__ wv,
                      const float*    __restrict__ in_proj_b,
                      _Float16*       __restrict__ ctx16) {
  __shared__ __align__(16) _Float16 sA[W_][H_ + 8];    // span tile, 32x768 f16
  __shared__ __align__(16) _Float16 sK[W_][HD_ + 8];   // per-head K tile
  __shared__ __align__(16) _Float16 sV[W_][HD_ + 8];   // per-head V tile
  __shared__ float sQ[H_];
  __shared__ float sAttn[W_];

  const int n    = blockIdx.x;       // span index
  const int b    = n / NS_;
  const int t    = threadIdx.x;
  const int lane = t & 31;
  const int wvid = t >> 5;

  const int start = span_ids[2 * n + 0];
  const int send  = span_ids[2 * n + 1];
  const int len   = send - start;

  for (int i = t; i < H_; i += 256) sQ[i] = qf[i];

  // gather span rows (masked), 16B vector moves
  for (int i = t; i < W_ * (H_ / 8); i += 256) {
    int w = i / (H_ / 8);
    int c = (i % (H_ / 8)) * 8;
    int pos = start + w;
    pos = pos < 0 ? 0 : (pos > SL_ - 1 ? SL_ - 1 : pos);
    float4 val = make_float4(0.f, 0.f, 0.f, 0.f);
    if (w < len)
      val = *(const float4*)&tokpe[((size_t)b * SL_ + pos) * H_ + c];
    *(float4*)&sA[w][c] = val;
  }
  __syncthreads();

  const int mt = wvid & 1;    // row tile (16 rows each)
  const int nt = wvid >> 1;   // column subtile within head (4 x 16)

  const int wrow  = mt * 16 + (lane & 15);
  const int kselA = (lane >> 4) * 16;
  const int kselB = (lane >> 4) * 16;

  for (int h = 0; h < NH_; ++h) {
    v8f accK = {};
    v8f accV = {};
    const int colw = h * HD_ + nt * 16 + (lane & 15);   // weight row (output col)
    const _Float16* wkp = wk + (size_t)colw * H_;
    const _Float16* wvp = wv + (size_t)colw * H_;

#pragma unroll
    for (int ks = 0; ks < H_; ks += 32) {
      if ((ks & 255) == 0) {
        __builtin_prefetch(wkp + ks + 256, 0, 0);
        __builtin_prefetch(wvp + ks + 256, 0, 0);
      }
      v16h a;  float4* ap  = (float4*)&a;
      ap[0] = *(const float4*)&sA[wrow][ks + kselA];
      ap[1] = *(const float4*)&sA[wrow][ks + kselA + 8];
      v16h bk; float4* bkp = (float4*)&bk;
      bkp[0] = *(const float4*)&wkp[ks + kselB];
      bkp[1] = *(const float4*)&wkp[ks + kselB + 8];
      v16h bv; float4* bvp = (float4*)&bv;
      bvp[0] = *(const float4*)&wvp[ks + kselB];
      bvp[1] = *(const float4*)&wvp[ks + kselB + 8];
      accK = __builtin_amdgcn_wmma_f32_16x16x32_f16(false, a, false, bk,
                                                    (short)0, accK, false, false);
      accV = __builtin_amdgcn_wmma_f32_16x16x32_f16(false, a, false, bv,
                                                    (short)0, accV, false, false);
    }

    // epilogue: +bias, f16 into LDS K/V tiles
    const int dcol = nt * 16 + (lane & 15);
    const float bK = in_proj_b[H_     + h * HD_ + dcol];
    const float bV = in_proj_b[2 * H_ + h * HD_ + dcol];
    const int rb = (lane >> 4) * 8;
#pragma unroll
    for (int r = 0; r < 8; ++r) {
      int w = mt * 16 + rb + r;
      sK[w][dcol] = (_Float16)(accK[r] + bK);
      sV[w][dcol] = (_Float16)(accV[r] + bV);
    }
    __syncthreads();

    // masked softmax over W (wave 0, one lane per position; wave32 shuffles)
    if (wvid == 0) {
      int w = lane;
      float s = 0.f;
#pragma unroll
      for (int d = 0; d < HD_; ++d) s += sQ[h * HD_ + d] * (float)sK[w][d];
      s *= 0.125f;                       // 1/sqrt(HD)
      if (w >= len) s = -1.0e9f;
      float mx = s;
      for (int off = 16; off; off >>= 1) mx = fmaxf(mx, __shfl_xor(mx, off, 32));
      float e = __expf(s - mx);
      float sum = e;
      for (int off = 16; off; off >>= 1) sum += __shfl_xor(sum, off, 32);
      sAttn[w] = e / sum;
    }
    __syncthreads();

    // ctx[h,d] = sum_w attn[w] * V[w,d]
    if (t < HD_) {
      float c = 0.f;
#pragma unroll
      for (int w = 0; w < W_; ++w) c += sAttn[w] * (float)sV[w][t];
      ctx16[(size_t)n * H_ + h * HD_ + t] = (_Float16)c;
    }
    __syncthreads();   // before next head reuses sK/sV
  }
}

// =====================================================================
// Generic WMMA GEMM:  out = A (MxK,f16) @ Wt^T (Wt is NcxK,f16) + bias
// optional: row-broadcast residual, elementwise residual, ReLU,
// f32 and/or f16 outputs.  64x64 block tile, 8 waves, BK=32.
// Tile movement: TDM (tensor_load_to_lds) when available, else
// cooperative b128 loads.
// =====================================================================
__global__ __launch_bounds__(256)
void gemm_bias_kernel(const _Float16* __restrict__ A,
                      const _Float16* __restrict__ Wt,
                      const float*    __restrict__ bias,
                      const float*    __restrict__ resRow,
                      const float*    __restrict__ resElem,
                      float*          __restrict__ outF32,
                      _Float16*       __restrict__ outF16,
                      int M, int Nc, int K, int relu) {
  __shared__ __align__(16) _Float16 sA2[64][32 + 8];   // 40-half rows: 64B + 16B pad
  __shared__ __align__(16) _Float16 sB2[64][32 + 8];

  const int t    = threadIdx.x;
  const int lane = t & 31;
  const int wvid = t >> 5;
  const int mt   = wvid & 3;    // 4 row tiles of 16
  const int nt   = wvid >> 2;   // 2 col halves of 32

  const int m0 = blockIdx.x * 64;
  const int n0 = blockIdx.y * 64;

  v8f acc0 = {};
  v8f acc1 = {};

  const int arow  = mt * 16 + (lane & 15);
  const int kselA = (lane >> 4) * 16;
  const int kselB = (lane >> 4) * 16;

#if HAVE_TDM_
  const unsigned int ldsA_off =
      (unsigned int)(uintptr_t)(&sA2[0][0]);
  const unsigned int ldsB_off =
      (unsigned int)(uintptr_t)(&sB2[0][0]);
#else
  const int ldrow = t >> 2;
  const int ldcol = (t & 3) << 3;
#endif

  for (int ks = 0; ks < K; ks += 32) {
#if HAVE_TDM_
    // one wave drives the DMA; TENSORcnt fence, then barrier releases all.
    if (wvid == 0) {
      tdm_load_tile_f16(ldsA_off, &A[(size_t)m0 * K + ks], 32u, 64u,
                        (unsigned int)K, 1);
      tdm_load_tile_f16(ldsB_off, &Wt[(size_t)n0 * K + ks], 32u, 64u,
                        (unsigned int)K, 1);
      __builtin_amdgcn_s_wait_tensorcnt(0);
    }
    __syncthreads();
#else
    *(float4*)&sA2[ldrow][ldcol] =
        *(const float4*)&A[(size_t)(m0 + ldrow) * K + ks + ldcol];
    *(float4*)&sB2[ldrow][ldcol] =
        *(const float4*)&Wt[(size_t)(n0 + ldrow) * K + ks + ldcol];
    __builtin_prefetch(&Wt[(size_t)(n0 + ldrow) * K + ks + 32 + ldcol], 0, 0);
    __syncthreads();
#endif

    v16h a; float4* ap = (float4*)&a;
    ap[0] = *(const float4*)&sA2[arow][kselA];
    ap[1] = *(const float4*)&sA2[arow][kselA + 8];

    {
      int ncol = nt * 32 + (lane & 15);
      v16h bf; float4* bp = (float4*)&bf;
      bp[0] = *(const float4*)&sB2[ncol][kselB];
      bp[1] = *(const float4*)&sB2[ncol][kselB + 8];
      acc0 = __builtin_amdgcn_wmma_f32_16x16x32_f16(false, a, false, bf,
                                                    (short)0, acc0, false, false);
    }
    {
      int ncol = nt * 32 + 16 + (lane & 15);
      v16h bf; float4* bp = (float4*)&bf;
      bp[0] = *(const float4*)&sB2[ncol][kselB];
      bp[1] = *(const float4*)&sB2[ncol][kselB + 8];
      acc1 = __builtin_amdgcn_wmma_f32_16x16x32_f16(false, a, false, bf,
                                                    (short)0, acc1, false, false);
    }
    __syncthreads();
  }

  // epilogue
  const int rb = (lane >> 4) * 8;
#pragma unroll
  for (int s = 0; s < 2; ++s) {
    const v8f acc = s ? acc1 : acc0;
    const int ncol = n0 + nt * 32 + s * 16 + (lane & 15);
    float bv = bias   ? bias[ncol]   : 0.f;
    float rr = resRow ? resRow[ncol] : 0.f;
#pragma unroll
    for (int r = 0; r < 8; ++r) {
      int m = m0 + mt * 16 + rb + r;
      float v = acc[r] + bv + rr;
      if (resElem) v += resElem[(size_t)m * Nc + ncol];
      if (relu)    v  = fmaxf(v, 0.f);
      size_t idx = (size_t)m * Nc + ncol;
      if (outF32) outF32[idx] = v;
      if (outF16) outF16[idx] = (_Float16)v;
    }
  }
}

// =====================================================================
// LayerNorm over H=768 per row; optional f32/f16 outputs; optional
// length mask (zero rows whose span length <= 0).
// =====================================================================
__global__ __launch_bounds__(256)
void layernorm_kernel(const float* __restrict__ in,
                      const float* __restrict__ g,
                      const float* __restrict__ bta,
                      float*       __restrict__ out32,
                      _Float16*    __restrict__ out16,
                      const int*   __restrict__ span_ids) {
  __shared__ float rs[8], rs2[8];
  __shared__ float smu, srs;
  const int n    = blockIdx.x;
  const int t    = threadIdx.x;
  const int lane = t & 31;
  const int wvid = t >> 5;

  float s = 0.f, s2 = 0.f;
  for (int i = t; i < H_; i += 256) {
    float v = in[(size_t)n * H_ + i];
    s += v; s2 += v * v;
  }
  for (int off = 16; off; off >>= 1) {
    s  += __shfl_xor(s,  off, 32);
    s2 += __shfl_xor(s2, off, 32);
  }
  if (lane == 0) { rs[wvid] = s; rs2[wvid] = s2; }
  __syncthreads();
  if (t == 0) {
    float S = 0.f, S2 = 0.f;
    for (int w = 0; w < 8; ++w) { S += rs[w]; S2 += rs2[w]; }
    float mu  = S / (float)H_;
    float var = S2 / (float)H_ - mu * mu;
    smu = mu;
    srs = rsqrtf(var + LN_EPS_);
  }
  __syncthreads();

  float mask = 1.f;
  if (span_ids) {
    int len = span_ids[2 * n + 1] - span_ids[2 * n];
    mask = (len > 0) ? 1.f : 0.f;
  }
  const float mu = smu, rstd = srs;
  for (int i = t; i < H_; i += 256) {
    float v = ((in[(size_t)n * H_ + i] - mu) * rstd * g[i] + bta[i]) * mask;
    if (out32) out32[(size_t)n * H_ + i] = v;
    if (out16) out16[(size_t)n * H_ + i] = (_Float16)v;
  }
}

// =====================================================================
// host-side orchestration
// =====================================================================
extern "C" void kernel_launch(void* const* d_in, const int* in_sizes, int n_in,
                              void* d_out, int out_size, void* d_ws, size_t ws_size,
                              hipStream_t stream) {
  const float* token_reps = (const float*)d_in[0];
  const int*   span_ids   = (const int*)  d_in[1];
  const float* q0         = (const float*)d_in[2];
  const float* in_proj_w  = (const float*)d_in[3];
  const float* in_proj_b  = (const float*)d_in[4];
  const float* out_proj_w = (const float*)d_in[5];
  const float* out_proj_b = (const float*)d_in[6];
  const float* ln_g       = (const float*)d_in[7];
  const float* ln_b       = (const float*)d_in[8];
  const float* ffn_w1     = (const float*)d_in[9];
  const float* ffn_b1     = (const float*)d_in[10];
  const float* ffn_w2     = (const float*)d_in[11];
  const float* ffn_b2     = (const float*)d_in[12];
  float* out = (float*)d_out;

  char* ws = (char*)d_ws;
  size_t off = 0;
  auto carve = [&](size_t bytes) -> char* {
    char* p = ws + off;
    off = (off + bytes + 255) & ~(size_t)255;
    return p;
  };

  _Float16* tokpe = (_Float16*)carve((size_t)B_ * SL_ * H_ * 2);
  _Float16* wk    = (_Float16*)carve((size_t)H_ * H_ * 2);
  _Float16* wv    = (_Float16*)carve((size_t)H_ * H_ * 2);
  _Float16* wo    = (_Float16*)carve((size_t)H_ * H_ * 2);
  _Float16* w1    = (_Float16*)carve((size_t)FFN_ * H_ * 2);
  _Float16* w2    = (_Float16*)carve((size_t)FFN_ * H_ * 2);
  float*    qf    = (float*)   carve((size_t)H_ * 4);
  _Float16* ctx16 = (_Float16*)carve((size_t)N_ * H_ * 2);
  float*    attno = (float*)   carve((size_t)N_ * H_ * 4);
  float*    x32   = (float*)   carve((size_t)N_ * H_ * 4);
  _Float16* x16   = (_Float16*)carve((size_t)N_ * H_ * 2);
  _Float16* h116  = (_Float16*)carve((size_t)N_ * FFN_ * 2);
  float*    y32   = (float*)   carve((size_t)N_ * H_ * 4);

  // 1. token + positional encoding -> f16
  prep_tokpe_kernel<<<2048, 256, 0, stream>>>(token_reps, tokpe, B_ * SL_ * H_);
  // 2. weights -> f16
  prep_weights_kernel<<<2048, 256, 0, stream>>>(in_proj_w, out_proj_w, ffn_w1,
                                                ffn_w2, wk, wv, wo, w1, w2);
  // 3. q projection
  qproj_kernel<<<3, 256, 0, stream>>>(in_proj_w, in_proj_b, q0, qf);
  // 4. fused gather + K/V WMMA projection + attention -> ctx
  span_attn_kernel<<<N_, 256, 0, stream>>>(tokpe, span_ids, qf, wk, wv,
                                           in_proj_b, ctx16);
  // 5. attn_out = ctx @ Wo^T + bo + q0 (row-broadcast residual)
  gemm_bias_kernel<<<dim3(N_ / 64, H_ / 64), 256, 0, stream>>>(
      ctx16, wo, out_proj_b, q0, nullptr, attno, nullptr, N_, H_, H_, 0);
  // 6. x = LN(attn_out)
  layernorm_kernel<<<N_, 256, 0, stream>>>(attno, ln_g, ln_b, x32, x16, nullptr);
  // 7. h1 = relu(x @ W1^T + b1)
  gemm_bias_kernel<<<dim3(N_ / 64, FFN_ / 64), 256, 0, stream>>>(
      x16, w1, ffn_b1, nullptr, nullptr, nullptr, h116, N_, FFN_, H_, 1);
  // 8. y = h1 @ W2^T + b2 + x
  gemm_bias_kernel<<<dim3(N_ / 64, H_ / 64), 256, 0, stream>>>(
      h116, w2, ffn_b2, nullptr, x32, y32, nullptr, N_, H_, FFN_, 0);
  // 9. out = mask(LN(y))
  layernorm_kernel<<<N_, 256, 0, stream>>>(y32, ln_g, ln_b, out, nullptr,
                                           span_ids);
  (void)in_sizes; (void)n_in; (void)out_size; (void)ws_size;
}
